// LSTM_72988674228703
// MI455X (gfx1250) — compile-verified, matrix operand order checked by reading, not run on hardware
//
#include <hip/hip_runtime.h>

#define H  512
#define BB 64
#define SS 1024
#define DD 512
#define GG 2048   // 4*H
#define NT 128    // GG/16 n-tiles
#define KT 16     // 512/32 k-tiles

typedef __bf16 bf16_t;
typedef __attribute__((ext_vector_type(16))) __bf16 v16bf;
typedef __attribute__((ext_vector_type(8)))  __bf16 v8bf;
typedef __attribute__((ext_vector_type(8)))  float  v8f;

// ---- workspace layout (bytes) ----
#define WFRAG_HALF  (2*2*128*16*512)               // 4,194,304 halfs = 8MB
#define XFRAG_HALF  (4*1024*16*512)                // 33,554,432 halfs = 64MB
#define XGFRAG_HALF (4*1024*128*256)               // 134,217,728 halfs = 256MB
#define XFRAG_OFF   ((size_t)WFRAG_HALF*2)
#define HFRAG_OFF   (XFRAG_OFF + (size_t)XFRAG_HALF*2)
#define XGFRAG_OFF  (HFRAG_OFF + (size_t)XFRAG_HALF*2)

// A/B 16-bit fragment K index for (lane-half hi, half-slot j)
__device__ __forceinline__ int kmap(int hi, int j) {
  return (j < 8) ? (hi*8 + j) : (16 + hi*8 + (j - 8));
}
__device__ __forceinline__ float sigmf(float x) { return 1.0f / (1.0f + __expf(-x)); }
__device__ __forceinline__ v8f wmma_bf16(v16bf a, v16bf b, v8f c) {
  return __builtin_amdgcn_wmma_f32_16x16x32_bf16(false, a, false, b, (short)0, c, false, false);
}

// ---------- one-time: W (L,4H,D+H) fp32 -> bf16 B-fragments [l][Wx/Wh][nt][kt][lane][16] ----------
__global__ __launch_bounds__(256) void wconv_kernel(const float* __restrict__ W,
                                                    bf16_t* __restrict__ wfrag) {
  long i = (long)blockIdx.x * blockDim.x + threadIdx.x;
  if (i >= (long)WFRAG_HALF) return;
  int j    = (int)(i & 15);
  int lane = (int)((i >> 4) & 31);
  int kt   = (int)((i >> 9) & 15);
  int nt   = (int)((i >> 13) & 127);
  int m    = (int)((i >> 20) & 1);     // 0 = Wx part, 1 = Wh part
  int l    = (int)(i >> 21);
  int hi = lane >> 4, n = lane & 15;
  int g = nt*16 + n;
  int d = kt*32 + kmap(hi, j);
  wfrag[i] = (bf16_t)W[((long)(l*GG + g))*1024 + (m ? 512 : 0) + d];
}

// ---------- one-time: x (B,S,D) fp32 -> bf16 A-fragments [bg][t][kt][lane][16] ----------
__global__ __launch_bounds__(256) void xconv_kernel(const float* __restrict__ x,
                                                    bf16_t* __restrict__ xfrag) {
  long i = (long)blockIdx.x * blockDim.x + threadIdx.x;
  if (i >= (long)XFRAG_HALF) return;
  int j    = (int)(i & 15);
  int lane = (int)((i >> 4) & 31);
  int kt   = (int)((i >> 9) & 15);
  int t    = (int)((i >> 13) & 1023);
  int bg   = (int)(i >> 23);
  int hi = lane >> 4, mrow = lane & 15;
  int b = bg*16 + mrow;
  int d = kt*32 + kmap(hi, j);
  xfrag[i] = (bf16_t)x[(((long)b*SS) + t)*DD + d];
}

// ---------- phase A: xg = in . Wx^T + bias ; 2 timesteps/block, pipelined operand loads ----------
__global__ __launch_bounds__(1024) void lstm_xgemm_kernel(
    const bf16_t* __restrict__ in_frag,   // [4][1024][16][512] A-fragments
    const bf16_t* __restrict__ wx_frag,   // [128][16][512] B-fragments
    const float*  __restrict__ bias,      // [2048]
    bf16_t* __restrict__ xg_frag)         // [4][1024][128][256] C-fragments
{
  const int tp = blockIdx.x;              // timestep pair 0..511
  const int bg = blockIdx.y;
  const int tid = threadIdx.x;
  const int w = tid >> 5, L = tid & 31, lm = L & 15;
  const int t0 = tp*2;

  v8f acc[2][4];
#pragma unroll
  for (int q = 0; q < 4; ++q) {
    const float bb = bias[(w*4 + q)*16 + lm];
    v8f a;
#pragma unroll
    for (int j = 0; j < 8; ++j) a[j] = bb;
    acc[0][q] = a;
    acc[1][q] = a;
  }

  const bf16_t* a0b = in_frag + (((long)bg*SS + t0)*KT)*512 + L*16;
  const bf16_t* a1b = a0b + (long)KT*512;          // t0+1
  const bf16_t* wb  = wx_frag + L*16;

  v16bf a0 = *(const v16bf*)a0b;
  v16bf a1 = *(const v16bf*)a1b;
  v16bf bcur[4];
#pragma unroll
  for (int q = 0; q < 4; ++q)
    bcur[q] = *(const v16bf*)(wb + ((long)(w*4 + q)*KT)*512);

#pragma unroll
  for (int kt = 0; kt < KT; ++kt) {
    v16bf an0, an1, bnext[4];
    if (kt < KT - 1) {                  // prefetch next k-tile's operands into registers
      an0 = *(const v16bf*)(a0b + (kt + 1)*512);
      an1 = *(const v16bf*)(a1b + (kt + 1)*512);
#pragma unroll
      for (int q = 0; q < 4; ++q)
        bnext[q] = *(const v16bf*)(wb + ((long)(w*4 + q)*KT + kt + 1)*512);
    }
#pragma unroll
    for (int q = 0; q < 4; ++q) {
      acc[0][q] = wmma_bf16(a0, bcur[q], acc[0][q]);
      acc[1][q] = wmma_bf16(a1, bcur[q], acc[1][q]);
    }
    if (kt < KT - 1) {
      a0 = an0; a1 = an1;
#pragma unroll
      for (int q = 0; q < 4; ++q) bcur[q] = bnext[q];
    }
  }

#pragma unroll
  for (int tt = 0; tt < 2; ++tt) {
    bf16_t* obase = xg_frag + (((long)bg*SS + t0 + tt)*NT)*256 + L*8;
#pragma unroll
    for (int q = 0; q < 4; ++q) {
      v8bf o;
#pragma unroll
      for (int j = 0; j < 8; ++j) o[j] = (bf16_t)acc[tt][q][j];
      *(v8bf*)(obase + (w*4 + q)*256) = o;
    }
  }
}

// ---------- phase B: sequential recurrence, one block (32 waves) per 16-batch group ----------
__global__ __launch_bounds__(1024) void lstm_recur_kernel(
    const bf16_t* __restrict__ xg_frag,   // [4][1024][128][256]
    const bf16_t* __restrict__ wh_frag,   // [128][16][512]
    const float*  __restrict__ h0,        // [B][H] (layer slice)
    const float*  __restrict__ c0,        // [B][H]
    bf16_t* __restrict__ hfrag_out,       // layer0: A-frag output; else null
    float*  __restrict__ out_f32,         // layer1: (B,S,H) output; else null
    float*  __restrict__ hn,              // [B][H]
    float*  __restrict__ cn)              // [B][H]
{
  __shared__ __align__(64) bf16_t ldsA[KT*512];   // h in A-fragment layout, 16KB

  const int bg = blockIdx.x, b0 = bg*16;
  const int tid = threadIdx.x;
  const int w = tid >> 5, L = tid & 31, hi = L >> 4, lm = L & 15;

  // this thread's fixed h-column and its A-fragment scatter slot (loop-invariant)
  const int col = w*16 + lm;
  const int kt2 = col >> 5, ko = col & 31;
  const int hi2 = (ko < 16) ? (ko >> 3) : ((ko - 16) >> 3);
  const int j2  = (ko < 16) ? (ko & 7) : (8 + ((ko - 16) & 7));

  float cst[8];
#pragma unroll
  for (int j = 0; j < 8; ++j) {
    const int m = j + hi*8;
    const int b = b0 + m;
    cst[j] = c0[(long)b*H + col];
    ldsA[kt2*512 + (hi2*16 + m)*16 + j2] = (bf16_t)h0[(long)b*H + col];
  }
  __syncthreads();

  const bf16_t* whb = wh_frag + L*16;

  for (int t = 0; t < SS; ++t) {
    // init accumulators from precomputed xg fragments (HBM stream)
    v8f acc[4];
    const bf16_t* xgb = xg_frag + (((long)bg*SS + t)*NT)*256 + L*8;
#pragma unroll
    for (int u = 0; u < 4; ++u) {
      const v8bf xv = *(const v8bf*)(xgb + (u*64 + w)*256);
      v8f a;
#pragma unroll
      for (int j = 0; j < 8; ++j) a[j] = (float)xv[j];
      acc[u] = a;
    }
    // prefetch next timestep's xg stream into cache (global_prefetch_b8)
    if (t + 1 < SS) {
      const bf16_t* xgn = xg_frag + (((long)bg*SS + t + 1)*NT)*256 + L*8;
#pragma unroll
      for (int u = 0; u < 4; ++u) __builtin_prefetch(xgn + (u*64 + w)*256, 0, 1);
    }

    // gates += h . Wh^T ; B double-buffered in registers to hide L2 latency
    v16bf bcur[4];
#pragma unroll
    for (int u = 0; u < 4; ++u)
      bcur[u] = *(const v16bf*)(whb + ((long)(u*64 + w)*KT)*512);
#pragma unroll
    for (int kt = 0; kt < KT; ++kt) {
      v16bf bnext[4];
      if (kt < KT - 1) {
#pragma unroll
        for (int u = 0; u < 4; ++u)
          bnext[u] = *(const v16bf*)(whb + ((long)(u*64 + w)*KT + kt + 1)*512);
      }
      const v16bf a = *(const v16bf*)(&ldsA[kt*512 + L*16]);
#pragma unroll
      for (int u = 0; u < 4; ++u) acc[u] = wmma_bf16(a, bcur[u], acc[u]);
      if (kt < KT - 1) {
#pragma unroll
        for (int u = 0; u < 4; ++u) bcur[u] = bnext[u];
      }
    }

    // elementwise cell update in fp32 (acc: 0=i, 1=f, 2=o, 3=g)
    float hv[8];
#pragma unroll
    for (int j = 0; j < 8; ++j) {
      const float ig = sigmf(acc[0][j]);
      const float fg = sigmf(acc[1][j]);
      const float og = sigmf(acc[2][j]);
      const float gg = tanhf(acc[3][j]);
      const float cnew = fg*cst[j] + ig*gg;
      cst[j] = cnew;
      hv[j] = og * tanhf(cnew);
    }
    __syncthreads();                       // everyone finished reading ldsA(t-1)
#pragma unroll
    for (int j = 0; j < 8; ++j) {          // scatter h(t) into A-fragment layout
      const int m = j + hi*8;
      ldsA[kt2*512 + (hi2*16 + m)*16 + j2] = (bf16_t)hv[j];
    }
    __syncthreads();                       // h(t) fragments visible

    if (hfrag_out) {                       // layer 0: emit h as next layer's A-fragments
      const int off = (w >> 1)*512 + (w & 1)*256 + L*8;
      *(v8bf*)(hfrag_out + (((long)bg*SS + t)*KT)*512 + off) =
          *(const v8bf*)(&ldsA[off]);
    }
    if (out_f32) {                         // layer 1: fp32 (B,S,H) output
#pragma unroll
      for (int j = 0; j < 8; ++j) {
        const int b = b0 + j + hi*8;
        out_f32[(((long)b*SS) + t)*H + col] = hv[j];
      }
    }
    if (t == SS - 1) {
#pragma unroll
      for (int j = 0; j < 8; ++j) {
        const int b = b0 + j + hi*8;
        hn[(long)b*H + col] = hv[j];
        cn[(long)b*H + col] = cst[j];
      }
    }
  }
}

extern "C" void kernel_launch(void* const* d_in, const int* in_sizes, int n_in,
                              void* d_out, int out_size, void* d_ws, size_t ws_size,
                              hipStream_t stream) {
  (void)in_sizes; (void)n_in; (void)out_size; (void)ws_size;
  const float* x  = (const float*)d_in[0];
  const float* W  = (const float*)d_in[1];
  const float* b  = (const float*)d_in[2];
  const float* h0 = (const float*)d_in[3];
  const float* c0 = (const float*)d_in[4];
  float* out = (float*)d_out;

  char* ws = (char*)d_ws;
  bf16_t* wfrag  = (bf16_t*)ws;
  bf16_t* xfrag  = (bf16_t*)(ws + XFRAG_OFF);
  bf16_t* hfrag  = (bf16_t*)(ws + HFRAG_OFF);
  bf16_t* xgfrag = (bf16_t*)(ws + XGFRAG_OFF);

  const long OUT0 = (long)BB*SS*H;     // 33,554,432 (end of `out`)
  const int  BH   = BB*H;              // 32,768
  const long WSEG = 128L*16*512;       // halfs per W matrix fragment block

  wconv_kernel<<<WFRAG_HALF/256, 256, 0, stream>>>(W, wfrag);
  xconv_kernel<<<XFRAG_HALF/256, 256, 0, stream>>>(x, xfrag);

  // layer 0
  lstm_xgemm_kernel<<<dim3(SS/2, 4), 1024, 0, stream>>>(xfrag, wfrag + 0*WSEG, b, xgfrag);
  lstm_recur_kernel<<<4, 1024, 0, stream>>>(xgfrag, wfrag + 1*WSEG, h0, c0,
      hfrag, (float*)nullptr, out + OUT0, out + OUT0 + 2*BH);
  // layer 1
  lstm_xgemm_kernel<<<dim3(SS/2, 4), 1024, 0, stream>>>(hfrag, wfrag + 2*WSEG, b + GG, xgfrag);
  lstm_recur_kernel<<<4, 1024, 0, stream>>>(xgfrag, wfrag + 3*WSEG, h0 + BH, c0 + BH,
      (bf16_t*)nullptr, out, out + OUT0 + BH, out + OUT0 + 2*BH + BH);
}
